// GCN_raw_mean_68968584839876
// MI455X (gfx1250) — matrix operand order repaired and checked by reading, not compiled, not run on hardware
//
#include <hip/hip_runtime.h>
#include <hip/hip_bf16.h>
#include <math.h>

typedef __attribute__((ext_vector_type(16))) _Float16 v16h;
typedef __attribute__((ext_vector_type(8)))  float    v8f;

#define EPSV 1e-5f

// ---------------------------------------------------------------- zero fill
__global__ void zero_f32(float* __restrict__ p, long long n) {
  long long i = blockIdx.x * (long long)blockDim.x + threadIdx.x;
  long long stride = gridDim.x * (long long)blockDim.x;
  for (; i < n; i += stride) p[i] = 0.0f;
}

// ---------------------------------------------------------------- weight prep
struct PrepArgs {
  const float *W1r, *b1, *W1s;
  const float *W2r, *b2, *W2s;
  const float *W3r, *b3, *W3s;
  const float *g1, *be1, *m1, *v1;
  const float *g2, *be2, *m2, *v2;
  const float *g3, *be3, *m3, *v3;
  _Float16 *Wc1, *Wc2, *Wc3;
  float *sc1, *sh1, *sc2, *sh2, *sc3, *sh3;
};

// Build stacked f16 weights Wc[k][o] = k<Cin ? W_rel[k][o] : W_root[k-Cin][o] (zero pad),
// and fold bias+batchnorm into per-channel scale/shift.
__global__ void prep_kernel(PrepArgs p) {
  int t = threadIdx.x;
  for (int i = t; i < 32 * 16; i += blockDim.x) {          // layer1: K_PAD=32, COUT=16, CIN=4
    int k = i / 16, o = i % 16;
    float v = (k < 4) ? p.W1r[k * 16 + o] : (k < 8) ? p.W1s[(k - 4) * 16 + o] : 0.0f;
    p.Wc1[i] = (_Float16)v;
  }
  for (int i = t; i < 32 * 32; i += blockDim.x) {          // layer2: K=32, COUT=32, CIN=16
    int k = i / 32, o = i % 32;
    float v = (k < 16) ? p.W2r[k * 32 + o] : p.W2s[(k - 16) * 32 + o];
    p.Wc2[i] = (_Float16)v;
  }
  for (int i = t; i < 64 * 64; i += blockDim.x) {          // layer3: K=64, COUT=64, CIN=32
    int k = i / 64, o = i % 64;
    float v = (k < 32) ? p.W3r[k * 64 + o] : p.W3s[(k - 32) * 64 + o];
    p.Wc3[i] = (_Float16)v;
  }
  for (int o = t; o < 16; o += blockDim.x) {
    float s = p.g1[o] * rsqrtf(p.v1[o] + EPSV);
    p.sc1[o] = s; p.sh1[o] = p.be1[o] + s * (p.b1[o] - p.m1[o]);
  }
  for (int o = t; o < 32; o += blockDim.x) {
    float s = p.g2[o] * rsqrtf(p.v2[o] + EPSV);
    p.sc2[o] = s; p.sh2[o] = p.be2[o] + s * (p.b2[o] - p.m2[o]);
  }
  for (int o = t; o < 64; o += blockDim.x) {
    float s = p.g3[o] * rsqrtf(p.v3[o] + EPSV);
    p.sc3[o] = s; p.sh3[o] = p.be3[o] + s * (p.b3[o] - p.m3[o]);
  }
}

// ---------------------------------------------------------------- edge scatter-add
// One thread per (edge, 4-channel group): float4 gather from h[src], f32 atomics into agg[dst].
template <int CIN>
__global__ void edge_agg(const int* __restrict__ ei, const float* __restrict__ ew,
                         const float* __restrict__ h, float* __restrict__ agg, int E) {
  constexpr int G = CIN / 4;
  long long tid = blockIdx.x * (long long)blockDim.x + threadIdx.x;
  long long total = (long long)E * G;
  long long stride = gridDim.x * (long long)blockDim.x;
  for (; tid < total; tid += stride) {
    int e = (int)(tid / G);
    int g = (int)(tid % G);
    int src = ei[e];
    int dst = ei[E + e];
    float w = ew[e];
    float4 v = reinterpret_cast<const float4*>(h + (size_t)src * CIN)[g];
    float* out = agg + (size_t)dst * CIN + (size_t)g * 4;
    atomicAdd(out + 0, w * v.x);
    atomicAdd(out + 1, w * v.y);
    atomicAdd(out + 2, w * v.z);
    atomicAdd(out + 3, w * v.w);
  }
}

// ---------------------------------------------------------------- WMMA node transform
// out[16xCOUT tile] = relu(scale * ([agg|h] @ [W_rel;W_root]) + shift)
// A fragment (16x32 f16): row M = lane%16; element e -> K = kf*32 + (e<8?e:e+8) + 8*(lane>=16)
//   => elements [0..7]  are the 8 consecutive K at  kf*32 +  0 + hi*8
//      elements [8..15] are the 8 consecutive K at  kf*32 + 16 + hi*8
// B fragment (32x16 f16): col N = lane%16; element e -> K = kf*32 + (e<8?e:e+8) + 8*(lane>=16)
// C/D (16x16 f32): element r -> M = r + 8*(lane>=16), N = lane%16
// Requires N % 16 == 0 (true here: 100000 = 6250*16).
template <int KP, int COUT, int CIN>
__global__ void node_wmma(const float* __restrict__ agg, const float* __restrict__ hin,
                          const _Float16* __restrict__ Wc, const float* __restrict__ scale,
                          const float* __restrict__ shift, float* __restrict__ hout,
                          int ntiles) {
  constexpr int KF = KP / 32;
  constexpr int NT = COUT / 16;
  const int lane = threadIdx.x & 31;
  const int wave = threadIdx.x >> 5;
  const int wavesPerBlock = blockDim.x >> 5;
  const int col0 = lane & 15;
  const int hi = lane >> 4;  // 0 for lanes 0-15, 1 for lanes 16-31

  // Load B fragments once per wave (weights reused across all tiles).
  v16h B[KF][NT];
#pragma unroll
  for (int kf = 0; kf < KF; kf++)
#pragma unroll
    for (int nt = 0; nt < NT; nt++)
#pragma unroll
      for (int e = 0; e < 16; e++) {
        int K = kf * 32 + (e < 8 ? e : e + 8) + hi * 8;
        B[kf][nt][e] = Wc[K * COUT + nt * 16 + col0];
      }

  int t = blockIdx.x * wavesPerBlock + wave;
  int tstride = gridDim.x * wavesPerBlock;
  for (; t < ntiles; t += tstride) {
    int node = t * 16 + col0;  // A-row handled by this lane
    const float* aggrow = agg + (size_t)node * CIN;
    const float* hrow = hin + (size_t)node * CIN;

    // Build A fragments with contiguous float4 loads (no divergent branches).
    v16h afrag[KF];
    if constexpr (CIN == 4) {
      // Valid K: 0..3 -> agg, 4..7 -> h, rest zero. Upper half-wave (hi=1)
      // covers K 8..15 / 24..31 which are all zero padding.
      float4 av = *reinterpret_cast<const float4*>(aggrow);
      float4 xv = *reinterpret_cast<const float4*>(hrow);
      float zf = (hi == 0) ? 1.0f : 0.0f;  // v_cndmask, no exec divergence
      v16h a;
      a[0] = (_Float16)(av.x * zf); a[1] = (_Float16)(av.y * zf);
      a[2] = (_Float16)(av.z * zf); a[3] = (_Float16)(av.w * zf);
      a[4] = (_Float16)(xv.x * zf); a[5] = (_Float16)(xv.y * zf);
      a[6] = (_Float16)(xv.z * zf); a[7] = (_Float16)(xv.w * zf);
#pragma unroll
      for (int e = 8; e < 16; e++) a[e] = (_Float16)0.0f;
      afrag[0] = a;
    } else {
      // CIN is a multiple of 16 -> each 16-wide K chunk maps entirely to agg or h,
      // decided at compile time after unrolling.
#pragma unroll
      for (int kf = 0; kf < KF; kf++) {
        v16h a;
#pragma unroll
        for (int c = 0; c < 2; c++) {
          const int kbase = kf * 32 + c * 16;  // constant after unroll
          const float* srcp = (kbase < CIN) ? (aggrow + kbase) : (hrow + (kbase - CIN));
          const float4* vp = reinterpret_cast<const float4*>(srcp + hi * 8);
          float4 p0 = vp[0];
          float4 p1 = vp[1];
          a[c * 8 + 0] = (_Float16)p0.x; a[c * 8 + 1] = (_Float16)p0.y;
          a[c * 8 + 2] = (_Float16)p0.z; a[c * 8 + 3] = (_Float16)p0.w;
          a[c * 8 + 4] = (_Float16)p1.x; a[c * 8 + 5] = (_Float16)p1.y;
          a[c * 8 + 6] = (_Float16)p1.z; a[c * 8 + 7] = (_Float16)p1.w;
        }
        afrag[kf] = a;
      }
    }

    v8f acc[NT];
    v8f zero = {};
#pragma unroll
    for (int nt = 0; nt < NT; nt++) acc[nt] = zero;

#pragma unroll
    for (int kf = 0; kf < KF; kf++)
#pragma unroll
      for (int nt = 0; nt < NT; nt++)
        acc[nt] = __builtin_amdgcn_wmma_f32_16x16x32_f16(
            false, afrag[kf], false, B[kf][nt], (short)0, acc[nt], false, false);

    // Fused bias+BN+ReLU epilogue. Base pointer + compile-time row offsets
    // (r*COUT) -> global_store_b32 with immediate offsets, no guards.
#pragma unroll
    for (int nt = 0; nt < NT; nt++) {
      int col = nt * 16 + col0;
      float sc = scale[col];
      float sh = shift[col];
      float* outp = hout + (size_t)(t * 16 + hi * 8) * COUT + col;
#pragma unroll
      for (int r = 0; r < 8; r++) {
        float v = acc[nt][r] * sc + sh;
        outp[(size_t)r * COUT] = v > 0.0f ? v : 0.0f;
      }
    }
  }
}

// ---------------------------------------------------------------- mean pool (segment sum)
__global__ void pool_kernel(const float* __restrict__ h3, const int* __restrict__ batch,
                            float* __restrict__ pool, float* __restrict__ counts, int N) {
  long long tid = blockIdx.x * (long long)blockDim.x + threadIdx.x;
  long long total = (long long)N * 64;
  long long stride = gridDim.x * (long long)blockDim.x;
  for (; tid < total; tid += stride) {
    int node = (int)(tid >> 6);
    int c = (int)(tid & 63);
    int g = batch[node];
    atomicAdd(&pool[(size_t)g * 64 + c], h3[tid]);
    if (c == 0) atomicAdd(&counts[g], 1.0f);
  }
}

// ---------------------------------------------------------------- MLP head
__global__ void head_kernel(const float* __restrict__ pool, const float* __restrict__ counts,
                            const float* __restrict__ Wfc1, const float* __restrict__ bfc1,
                            const float* __restrict__ Wfc4, const float* __restrict__ bfc4,
                            const float* __restrict__ gf, const float* __restrict__ bef,
                            const float* __restrict__ mf, const float* __restrict__ vf,
                            float* __restrict__ out, int NG) {
  int g = blockIdx.x * blockDim.x + threadIdx.x;
  if (g >= NG) return;
  float cnt = counts[g];
  cnt = cnt > 1.0f ? cnt : 1.0f;
  float mean[64];
#pragma unroll
  for (int i = 0; i < 64; i++) mean[i] = pool[(size_t)g * 64 + i] / cnt;
  float r[16];
#pragma unroll
  for (int o = 0; o < 16; o++) {
    float s = bfc1[o];
#pragma unroll
    for (int i = 0; i < 64; i++) s += mean[i] * Wfc1[i * 16 + o];
    float sc = gf[o] * rsqrtf(vf[o] + EPSV);
    float v = sc * (s - mf[o]) + bef[o];
    r[o] = v > 0.0f ? v : 0.0f;
  }
  float l0 = bfc4[0], l1 = bfc4[1];
#pragma unroll
  for (int o = 0; o < 16; o++) {
    l0 += r[o] * Wfc4[o * 2 + 0];
    l1 += r[o] * Wfc4[o * 2 + 1];
  }
  float mx = l0 > l1 ? l0 : l1;
  float lse = mx + logf(expf(l0 - mx) + expf(l1 - mx));
  out[g * 2 + 0] = l0 - lse;
  out[g * 2 + 1] = l1 - lse;
}

// ---------------------------------------------------------------- launcher
extern "C" void kernel_launch(void* const* d_in, const int* in_sizes, int n_in,
                              void* d_out, int out_size, void* d_ws, size_t ws_size,
                              hipStream_t stream) {
  // Input order: x, edge_index, edge_weight, batch, then params in dict order.
  const float* x    = (const float*)d_in[0];
  const int*   ei   = (const int*)d_in[1];
  const float* ew   = (const float*)d_in[2];
  const int*   batch = (const int*)d_in[3];
  const float* W1r = (const float*)d_in[4];
  const float* b1  = (const float*)d_in[5];
  const float* W1s = (const float*)d_in[6];
  const float* W2r = (const float*)d_in[7];
  const float* b2  = (const float*)d_in[8];
  const float* W2s = (const float*)d_in[9];
  const float* W3r = (const float*)d_in[10];
  const float* b3  = (const float*)d_in[11];
  const float* W3s = (const float*)d_in[12];
  const float* Wfc1 = (const float*)d_in[13];
  const float* bfc1 = (const float*)d_in[14];
  const float* Wfc4 = (const float*)d_in[15];
  const float* bfc4 = (const float*)d_in[16];
  const float* g1 = (const float*)d_in[17], *be1 = (const float*)d_in[18],
             * m1 = (const float*)d_in[19], *v1  = (const float*)d_in[20];
  const float* g2 = (const float*)d_in[21], *be2 = (const float*)d_in[22],
             * m2 = (const float*)d_in[23], *v2  = (const float*)d_in[24];
  const float* g3 = (const float*)d_in[25], *be3 = (const float*)d_in[26],
             * m3 = (const float*)d_in[27], *v3  = (const float*)d_in[28];
  const float* gf = (const float*)d_in[29], *bef = (const float*)d_in[30],
             * mf = (const float*)d_in[31], *vf  = (const float*)d_in[32];

  const int E  = in_sizes[2];       // 3,200,000 edges
  const int N  = in_sizes[3];       // 100,000 nodes (multiple of 16)
  const int NG = out_size / 2;      // 512 graphs

  // Workspace carve-up (256B aligned slabs).
  char* wp = (char*)d_ws;
  auto alloc = [&](size_t bytes) -> char* {
    char* p = wp;
    wp += (bytes + 255) & ~(size_t)255;
    return p;
  };
  float*    agg   = (float*)alloc(sizeof(float) * (size_t)N * 32);
  float*    h1    = (float*)alloc(sizeof(float) * (size_t)N * 16);
  float*    h2    = (float*)alloc(sizeof(float) * (size_t)N * 32);
  float*    h3    = (float*)alloc(sizeof(float) * (size_t)N * 64);
  float*    pool  = (float*)alloc(sizeof(float) * (size_t)NG * 64);
  float*    cnts  = (float*)alloc(sizeof(float) * (size_t)NG);
  _Float16* Wc1   = (_Float16*)alloc(sizeof(_Float16) * 32 * 16);
  _Float16* Wc2   = (_Float16*)alloc(sizeof(_Float16) * 32 * 32);
  _Float16* Wc3   = (_Float16*)alloc(sizeof(_Float16) * 64 * 64);
  float*    sc1   = (float*)alloc(sizeof(float) * 16);
  float*    sh1   = (float*)alloc(sizeof(float) * 16);
  float*    sc2   = (float*)alloc(sizeof(float) * 32);
  float*    sh2   = (float*)alloc(sizeof(float) * 32);
  float*    sc3   = (float*)alloc(sizeof(float) * 64);
  float*    sh3   = (float*)alloc(sizeof(float) * 64);
  (void)ws_size; (void)n_in;

  // 0) fold weights/bias/BN
  PrepArgs pa;
  pa.W1r = W1r; pa.b1 = b1; pa.W1s = W1s;
  pa.W2r = W2r; pa.b2 = b2; pa.W2s = W2s;
  pa.W3r = W3r; pa.b3 = b3; pa.W3s = W3s;
  pa.g1 = g1; pa.be1 = be1; pa.m1 = m1; pa.v1 = v1;
  pa.g2 = g2; pa.be2 = be2; pa.m2 = m2; pa.v2 = v2;
  pa.g3 = g3; pa.be3 = be3; pa.m3 = m3; pa.v3 = v3;
  pa.Wc1 = Wc1; pa.Wc2 = Wc2; pa.Wc3 = Wc3;
  pa.sc1 = sc1; pa.sh1 = sh1; pa.sc2 = sc2; pa.sh2 = sh2; pa.sc3 = sc3; pa.sh3 = sh3;
  prep_kernel<<<1, 256, 0, stream>>>(pa);

  const int ntiles = N / 16;                // exact (N % 16 == 0)
  const int wmmaBlocks = (ntiles + 7) / 8;  // 8 waves of 32 per 256-thread block

  // 1) layer 1: agg = scatter(w * x[src]); h1 = relu(bn(wmma([agg|x] @ Wc1)))
  {
    long long nz = (long long)N * 4;
    zero_f32<<<(int)((nz + 255) / 256), 256, 0, stream>>>(agg, nz);
    long long tw = (long long)E * 1;
    edge_agg<4><<<(int)((tw + 255) / 256), 256, 0, stream>>>(ei, ew, x, agg, E);
    node_wmma<32, 16, 4><<<wmmaBlocks, 256, 0, stream>>>(agg, x, Wc1, sc1, sh1, h1, ntiles);
  }
  // 2) layer 2
  {
    long long nz = (long long)N * 16;
    zero_f32<<<(int)((nz + 255) / 256), 256, 0, stream>>>(agg, nz);
    long long tw = (long long)E * 4;
    edge_agg<16><<<(int)((tw + 255) / 256), 256, 0, stream>>>(ei, ew, h1, agg, E);
    node_wmma<32, 32, 16><<<wmmaBlocks, 256, 0, stream>>>(agg, h1, Wc2, sc2, sh2, h2, ntiles);
  }
  // 3) layer 3
  {
    long long nz = (long long)N * 32;
    zero_f32<<<(int)((nz + 255) / 256), 256, 0, stream>>>(agg, nz);
    long long tw = (long long)E * 8;
    edge_agg<32><<<(int)((tw + 255) / 256), 256, 0, stream>>>(ei, ew, h2, agg, E);
    node_wmma<64, 64, 32><<<wmmaBlocks, 256, 0, stream>>>(agg, h2, Wc3, sc3, sh3, h3, ntiles);
  }
  // 4) global mean pool + head
  {
    zero_f32<<<(int)(((long long)NG * 64 + 255) / 256), 256, 0, stream>>>(pool, (long long)NG * 64);
    zero_f32<<<1, 256, 0, stream>>>(cnts, NG);
    long long tw = (long long)N * 64;
    pool_kernel<<<(int)((tw + 255) / 256), 256, 0, stream>>>(h3, batch, pool, cnts, N);
    head_kernel<<<(NG + 255) / 256, 256, 0, stream>>>(pool, cnts, Wfc1, bfc1, Wfc4, bfc4,
                                                      gf, bef, mf, vf, (float*)d_out, NG);
  }
}
